// ResidualMambaBlock_62182536511817
// MI455X (gfx1250) — matrix operand order repaired and checked by reading, not compiled, not run on hardware
//
#include <hip/hip_runtime.h>
#include <hip/hip_bf16.h>
#include <cstdint>

// ---------------------------------------------------------------------------
// Mamba block for gfx1250 (MI455X): double-buffered TDM-staged WMMA f16 GEMMs
// + f32 selective scan.
// B=8, L=1024, D_MODEL=512, D_INNER=1024, D_STATE=16, D_CONV=4, DT_RANK=32
// ---------------------------------------------------------------------------

#define BATCH    8
#define SEQ      1024
#define DMODEL   512
#define DINNER   1024
#define DSTATE   16
#define DCONV    4
#define DTRANK   32
#define MTOK     (BATCH * SEQ)        // 8192 token rows

typedef __attribute__((ext_vector_type(16))) _Float16      v16h;
typedef __attribute__((ext_vector_type(8)))  _Float16      v8h;
typedef __attribute__((ext_vector_type(8)))  float         v8f;
typedef __attribute__((ext_vector_type(4)))  unsigned int  v4u;
typedef __attribute__((ext_vector_type(8)))  int           v8i;
typedef __attribute__((ext_vector_type(4)))  int           v4i;

// ---------------------------------------------------------------------------
// f32 -> f16 cast (weights)
// ---------------------------------------------------------------------------
__global__ void cast_f32_to_f16_kernel(const float* __restrict__ src,
                                       _Float16* __restrict__ dst, int n) {
    int i = blockIdx.x * blockDim.x + threadIdx.x;
    if (i < n) dst[i] = (_Float16)src[i];
}

// ---------------------------------------------------------------------------
// LayerNorm over 512 features + cast to f16. One block (256 thr) per row.
// ---------------------------------------------------------------------------
__global__ void layernorm_cast_kernel(const float* __restrict__ x,
                                      const float* __restrict__ w,
                                      const float* __restrict__ b,
                                      _Float16* __restrict__ out) {
    __shared__ float s1[256];
    __shared__ float s2[256];
    const int m = blockIdx.x;
    const int t = threadIdx.x;
    const float* row = x + (size_t)m * DMODEL;
    float a0 = row[t];
    float a1 = row[t + 256];
    s1[t] = a0 + a1;
    s2[t] = a0 * a0 + a1 * a1;
    __syncthreads();
    for (int off = 128; off > 0; off >>= 1) {
        if (t < off) { s1[t] += s1[t + off]; s2[t] += s2[t + off]; }
        __syncthreads();
    }
    float mu  = s1[0] * (1.0f / DMODEL);
    float var = s2[0] * (1.0f / DMODEL) - mu * mu;
    float rs  = rsqrtf(var + 1e-5f);
    out[(size_t)m * DMODEL + t]       = (_Float16)((a0 - mu) * rs * w[t]       + b[t]);
    out[(size_t)m * DMODEL + t + 256] = (_Float16)((a1 - mu) * rs * w[t + 256] + b[t + 256]);
}

// ---------------------------------------------------------------------------
// Tensor Data Mover: DMA a 2D tile (rows x tile_k halves, row pitch in halves)
// from global memory into LDS at lds_off. D# per CDNA5 ISA 8.3/8.4:
//   group0: count=1 | lds_addr | global_addr[56:0] | type=2
//   group1: data_size=2B, tensor_dim = tile_dim (tile fully in-bounds),
//           tile_dim0=tile_k, tile_dim1=rows, tensor_dim0_stride=row pitch
// 6-arg builtin (clang-23 / therock-10.0 form), confirmed by this toolchain.
// ---------------------------------------------------------------------------
__device__ __forceinline__ void tdm_load_2d_to_lds(unsigned lds_off,
                                                   const _Float16* gptr,
                                                   unsigned tile_k_halves,
                                                   unsigned rows,
                                                   unsigned pitch_halves) {
    unsigned long long ga = (unsigned long long)(uintptr_t)gptr;
    v4u g0;
    g0[0] = 1u;                                             // count=1, user mode
    g0[1] = lds_off;                                        // LDS byte address
    g0[2] = (unsigned)(ga & 0xFFFFFFFFu);                   // global_addr[31:0]
    g0[3] = (unsigned)((ga >> 32) & 0x01FFFFFFu) | (2u << 30);  // [56:32] | type=2

    v8i g1;
    g1[0] = (int)(1u << 16);                                // wg_mask=0, data_size=2B
    g1[1] = (int)((tile_k_halves & 0xFFFFu) << 16);         // tensor_dim0[15:0]
    g1[2] = (int)((tile_k_halves >> 16) | ((rows & 0xFFFFu) << 16)); // td0 hi | td1 lo
    g1[3] = (int)((rows >> 16) | (tile_k_halves << 16));    // td1 hi | tile_dim0
    g1[4] = (int)rows;                                      // tile_dim1 | tile_dim2=0
    g1[5] = (int)pitch_halves;                              // tensor_dim0_stride lo
    g1[6] = 0;                                              // stride hi | stride1 lo
    g1[7] = 0;
    v4i z4 = {0, 0, 0, 0};
    v8i z8 = {0, 0, 0, 0, 0, 0, 0, 0};
    __builtin_amdgcn_tensor_load_to_lds(g0, g1, z4, z4, z8, 0);
}

// ---------------------------------------------------------------------------
// WMMA operand builders (wave32 layouts per CDNA5 ISA 7.12.2).
// A 16x32 f16: lane row = lane&15; two 8-half chunks at K=(lane>>4)*8, +16.
// B 32x16 f16: lane col = lane&15; two 8-half chunks at K=(lane>>4)*8, +16.
// ---------------------------------------------------------------------------
__device__ __forceinline__ v16h load_a_tile(const _Float16* __restrict__ A,
                                            int lda, int row, int k, int lane) {
    const int kh = (lane >> 4) * 8;
    const _Float16* p = A + (size_t)row * lda + k + kh;
    v8h lo = *reinterpret_cast<const v8h*>(p);
    v8h hi = *reinterpret_cast<const v8h*>(p + 16);
    v16h r;
#pragma unroll
    for (int i = 0; i < 8; ++i) { r[i] = lo[i]; r[i + 8] = hi[i]; }
    return r;
}

__device__ __forceinline__ v16h load_b_tile_lds(const _Float16* __restrict__ bsm,
                                                int n_in_chunk, int k_local, int lane) {
    // LDS chunk layout: 64 rows x 128 halves, row pitch 128 halves (256B)
    const int row = n_in_chunk + (lane & 15);
    const int kb  = k_local + (lane >> 4) * 8;
    const _Float16* p = bsm + row * 128 + kb;
    v8h lo = *reinterpret_cast<const v8h*>(p);
    v8h hi = *reinterpret_cast<const v8h*>(p + 16);
    v16h r;
#pragma unroll
    for (int i = 0; i < 8; ++i) { r[i] = lo[i]; r[i + 8] = hi[i]; }
    return r;
}

// ---------------------------------------------------------------------------
// Blocked WMMA GEMM:  C[M,N] = A[M,K] * W[N,K]^T  (row-major, K contiguous)
// Block = 128 threads = 4 waves stacked along M; wave computes 32x64 output
// (2 M-tiles x 4 N-tiles, 8 f32 accumulators). The 64-row x 128-half B chunk
// shared by the block is DMA'd global->LDS by wave 0 with the Tensor Data
// Mover, DOUBLE-BUFFERED: the DMA for chunk c+1 is issued before waiting for
// chunk c (s_wait_tensorcnt 1), so the tensor engine overlaps WMMA compute.
// The end-of-chunk barrier closes the WAR hazard on the ping-pong buffer.
// A-tiles are direct pipelined global loads. grid = (N/64, M/128). K%128==0.
// ---------------------------------------------------------------------------
template <bool F16OUT, bool RESID>
__global__ __launch_bounds__(128)
void wmma_gemm_tdm_kernel(const _Float16* __restrict__ A,
                          const _Float16* __restrict__ W,
                          float* __restrict__ Cf,
                          _Float16* __restrict__ Ch,
                          const float* __restrict__ resid,
                          int M, int N, int K) {
    __shared__ _Float16 bsm[2][64 * 128];              // 2 x 16 KB B chunks
    const int lane  = threadIdx.x & 31;
    const int wave  = threadIdx.x >> 5;
    const int mBase = blockIdx.y * 128 + wave * 32;
    const int nBase = blockIdx.x * 64;
    const int nChunks = K >> 7;

    v8f acc[2][4] = {};

    if (wave == 0) {
        tdm_load_2d_to_lds((unsigned)(uintptr_t)&bsm[0][0],
                           W + (size_t)nBase * K,
                           /*tile_k=*/128, /*rows=*/64, /*pitch=*/(unsigned)K);
    }

    for (int c = 0; c < nChunks; ++c) {
        const int k0 = c << 7;
        if (wave == 0) {
            if (c + 1 < nChunks) {
                tdm_load_2d_to_lds((unsigned)(uintptr_t)&bsm[(c + 1) & 1][0],
                                   W + (size_t)nBase * K + k0 + 128,
                                   /*tile_k=*/128, /*rows=*/64, /*pitch=*/(unsigned)K);
                __builtin_amdgcn_s_wait_tensorcnt(1);   // chunk c landed
            } else {
                __builtin_amdgcn_s_wait_tensorcnt(0);   // final chunk landed
            }
        }
        __syncthreads();
        const _Float16* bcur = &bsm[c & 1][0];

#pragma unroll
        for (int kk = 0; kk < 4; ++kk) {
            const int k = k0 + kk * 32;
            v16h a0 = load_a_tile(A, K, mBase + (lane & 15),      k, lane);
            v16h a1 = load_a_tile(A, K, mBase + 16 + (lane & 15), k, lane);
            __builtin_prefetch(A + (size_t)(mBase + (lane & 15)) * K + k + 128, 0, 1);
#pragma unroll
            for (int ni = 0; ni < 4; ++ni) {
                v16h b = load_b_tile_lds(bcur, ni * 16, kk * 32, lane);
                acc[0][ni] = __builtin_amdgcn_wmma_f32_16x16x32_f16(
                    false, a0, false, b, (short)0, acc[0][ni], false, false);
                acc[1][ni] = __builtin_amdgcn_wmma_f32_16x16x32_f16(
                    false, a1, false, b, (short)0, acc[1][ni], false, false);
            }
        }
        __syncthreads();   // readers done before this buffer is re-filled
    }

    const int n  = lane & 15;
    const int mo = (lane >> 4) * 8;
#pragma unroll
    for (int mi = 0; mi < 2; ++mi) {
#pragma unroll
        for (int ni = 0; ni < 4; ++ni) {
#pragma unroll
            for (int r = 0; r < 8; ++r) {
                size_t idx = (size_t)(mBase + mi * 16 + mo + r) * N + nBase + ni * 16 + n;
                float v = acc[mi][ni][r];
                if (RESID) v += resid[idx];
                if (F16OUT) Ch[idx] = (_Float16)v;
                else        Cf[idx] = v;
            }
        }
    }
}

// ---------------------------------------------------------------------------
// Causal depthwise conv (4 taps) over time + SiLU. u = xz[:, :, 0:1024].
// ---------------------------------------------------------------------------
__global__ void conv_silu_kernel(const _Float16* __restrict__ xz,
                                 const float* __restrict__ conv_w,
                                 const float* __restrict__ conv_b,
                                 float* __restrict__ u_act,
                                 _Float16* __restrict__ u_act_h) {
    int idx = blockIdx.x * blockDim.x + threadIdx.x;
    if (idx >= MTOK * DINNER) return;
    const int d = idx & (DINNER - 1);
    const int m = idx >> 10;            // token row = b*L + t
    const int t = m & (SEQ - 1);

    float acc = conv_b[d];
#pragma unroll
    for (int j = 0; j < DCONV; ++j) {
        int tt = t - (DCONV - 1) + j;
        if (tt >= 0) {
            float uv = (float)xz[(size_t)(m - (DCONV - 1) + j) * (2 * DINNER) + d];
            acc += conv_w[d * DCONV + j] * uv;
        }
    }
    float s = acc / (1.0f + expf(-acc));   // SiLU
    u_act[idx]   = s;
    u_act_h[idx] = (_Float16)s;
}

// ---------------------------------------------------------------------------
// Selective scan. One thread per (b, d) channel; h[16] and A-row in registers.
// Block = 64 threads (64 channels); grid = B * (DINNER/64) = 128 blocks.
// Fuses dt_proj dot + softplus, state update, C-contraction, D-skip, SiLU(z).
// ---------------------------------------------------------------------------
__global__ __launch_bounds__(64)
void scan_kernel(const float* __restrict__ x_dbl,     // [MTOK,64]
                 const float* __restrict__ dt_proj_w, // [DINNER,32]
                 const float* __restrict__ dt_proj_b, // [DINNER]
                 const float* __restrict__ A_log,     // [DINNER,16]
                 const float* __restrict__ D_param,   // [DINNER]
                 const float* __restrict__ u_act,     // [MTOK,DINNER]
                 const _Float16* __restrict__ xz,     // [MTOK,2*DINNER]
                 _Float16* __restrict__ y_h) {        // [MTOK,DINNER]
    __shared__ float sdbl[64];
    const int bIdx = blockIdx.x >> 4;                       // batch
    const int d    = ((blockIdx.x & 15) << 6) + threadIdx.x;

    float Arow[DSTATE], h[DSTATE], dtw[DTRANK];
#pragma unroll
    for (int s = 0; s < DSTATE; ++s) {
        Arow[s] = -expf(A_log[d * DSTATE + s]);
        h[s] = 0.0f;
    }
#pragma unroll
    for (int r = 0; r < DTRANK; ++r) dtw[r] = dt_proj_w[d * DTRANK + r];
    const float dtb = dt_proj_b[d];
    const float Dp  = D_param[d];

    for (int t = 0; t < SEQ; ++t) {
        const int m = bIdx * SEQ + t;
        sdbl[threadIdx.x] = x_dbl[(size_t)m * 64 + threadIdx.x];
        __syncthreads();

        float dtv = dtb;
#pragma unroll
        for (int r = 0; r < DTRANK; ++r) dtv += sdbl[r] * dtw[r];
        float dt = (dtv > 20.0f) ? dtv : log1pf(expf(dtv));   // softplus

        const float u  = u_act[(size_t)m * DINNER + d];
        const float du = dt * u;
        float y = 0.0f;
#pragma unroll
        for (int s = 0; s < DSTATE; ++s) {
            h[s] = expf(dt * Arow[s]) * h[s] + du * sdbl[32 + s];
            y += h[s] * sdbl[48 + s];
        }
        y += Dp * u;
        float z = (float)xz[(size_t)m * (2 * DINNER) + DINNER + d];
        y *= z / (1.0f + expf(-z));                            // y * SiLU(z)
        y_h[(size_t)m * DINNER + d] = (_Float16)y;
        __syncthreads();
    }
}

// ---------------------------------------------------------------------------
// Host launcher
// ---------------------------------------------------------------------------
extern "C" void kernel_launch(void* const* d_in, const int* in_sizes, int n_in,
                              void* d_out, int out_size, void* d_ws, size_t ws_size,
                              hipStream_t stream) {
    (void)in_sizes; (void)n_in; (void)out_size; (void)ws_size;

    const float* x         = (const float*)d_in[0];   // [B,L,512]
    const float* norm_w    = (const float*)d_in[1];
    const float* norm_b    = (const float*)d_in[2];
    const float* in_proj_w = (const float*)d_in[3];   // [2048,512]
    const float* conv_w    = (const float*)d_in[4];   // [1024,1,4]
    const float* conv_b    = (const float*)d_in[5];
    const float* x_proj_w  = (const float*)d_in[6];   // [64,1024]
    const float* dt_proj_w = (const float*)d_in[7];   // [1024,32]
    const float* dt_proj_b = (const float*)d_in[8];
    const float* A_log     = (const float*)d_in[9];   // [1024,16]
    const float* D_param   = (const float*)d_in[10];
    const float* out_proj_w= (const float*)d_in[11];  // [512,1024]
    float* out = (float*)d_out;

    // ---- workspace carve-up (all offsets 256B-aligned by construction) ----
    char* ws = (char*)d_ws;
    size_t off = 0;
    auto take = [&](size_t bytes) { char* p = ws + off; off += (bytes + 255) & ~(size_t)255; return p; };
    _Float16* xz_h    = (_Float16*)take((size_t)MTOK * 2 * DINNER * 2);  // 32 MB
    float*    u_act   = (float*)   take((size_t)MTOK * DINNER * 4);      // 32 MB
    _Float16* u_act_h = (_Float16*)take((size_t)MTOK * DINNER * 2);      // 16 MB
    _Float16* xn_h    = (_Float16*)take((size_t)MTOK * DMODEL * 2);      // 8 MB
    float*    x_dbl   = (float*)   take((size_t)MTOK * 64 * 4);          // 2 MB
    _Float16* y_h     = (_Float16*)take((size_t)MTOK * DINNER * 2);      // 16 MB
    _Float16* inw_h   = (_Float16*)take((size_t)2 * DINNER * DMODEL * 2);
    _Float16* xpw_h   = (_Float16*)take((size_t)(DTRANK + 2 * DSTATE) * DINNER * 2);
    _Float16* opw_h   = (_Float16*)take((size_t)DMODEL * DINNER * 2);

    // 1) cast weights to f16
    {
        int n1 = 2 * DINNER * DMODEL;
        cast_f32_to_f16_kernel<<<(n1 + 255) / 256, 256, 0, stream>>>(in_proj_w, inw_h, n1);
        int n2 = (DTRANK + 2 * DSTATE) * DINNER;
        cast_f32_to_f16_kernel<<<(n2 + 255) / 256, 256, 0, stream>>>(x_proj_w, xpw_h, n2);
        int n3 = DMODEL * DINNER;
        cast_f32_to_f16_kernel<<<(n3 + 255) / 256, 256, 0, stream>>>(out_proj_w, opw_h, n3);
    }

    // 2) LayerNorm + cast
    layernorm_cast_kernel<<<MTOK, 256, 0, stream>>>(x, norm_w, norm_b, xn_h);

    // 3) in_proj GEMM: xz[8192,2048] = xn[8192,512] * W[2048,512]^T  (f16 out)
    {
        dim3 grid((2 * DINNER) / 64, MTOK / 128);
        wmma_gemm_tdm_kernel<true, false><<<grid, 128, 0, stream>>>(
            xn_h, inw_h, nullptr, xz_h, nullptr, MTOK, 2 * DINNER, DMODEL);
    }

    // 4) causal depthwise conv + SiLU
    {
        int n = MTOK * DINNER;
        conv_silu_kernel<<<(n + 255) / 256, 256, 0, stream>>>(xz_h, conv_w, conv_b, u_act, u_act_h);
    }

    // 5) x_proj GEMM: x_dbl[8192,64] = u[8192,1024] * W[64,1024]^T  (f32 out)
    {
        dim3 grid(64 / 64, MTOK / 128);
        wmma_gemm_tdm_kernel<false, false><<<grid, 128, 0, stream>>>(
            u_act_h, xpw_h, x_dbl, nullptr, nullptr, MTOK, DTRANK + 2 * DSTATE, DINNER);
    }

    // 6) selective scan (fused dt_proj + softplus + recurrence + gate)
    scan_kernel<<<BATCH * (DINNER / 64), 64, 0, stream>>>(
        x_dbl, dt_proj_w, dt_proj_b, A_log, D_param, u_act, xz_h, y_h);

    // 7) out_proj GEMM + residual: out[8192,512] = x + y[8192,1024] * W[512,1024]^T
    {
        dim3 grid(DMODEL / 64, MTOK / 128);
        wmma_gemm_tdm_kernel<false, true><<<grid, 128, 0, stream>>>(
            y_h, opw_h, out, nullptr, x, MTOK, DMODEL, DINNER);
    }
}
